// LSTM_Encoder_8349416423908
// MI455X (gfx1250) — matrix operand order repaired and checked by reading, not compiled, not run on hardware
//
#include <hip/hip_runtime.h>
#include <hip/hip_bf16.h>

typedef _Float16 half_t;
typedef __attribute__((ext_vector_type(16))) _Float16 v16h;
typedef __attribute__((ext_vector_type(8)))  _Float16 v8h;
typedef __attribute__((ext_vector_type(8)))  float    v8f;

#define B_  64
#define S_  1024
#define I_  300
#define IP_ 320      // I padded to multiple of 32 (K of wmma f16)
#define H_  512
#define G_  2048     // 4*H gates

union V16U { v16h v; v8h h[2]; };

// ---- fragment loaders: contiguous k order (k = 16*(lane/16) + s), same as packed B ----
// A (16x32, f16): lanes 0-15 rows M=0..15 kbase=0; lanes 16-31 same rows kbase=16.
// halves 0..15 -> K = kbase+0..15  (two contiguous v8h)
static __device__ __forceinline__ v16h load_a_frag(const half_t* __restrict__ base,
                                                   long rowStride, int lane) {
  const int row = lane & 15;
  const int kb  = (lane >> 4) << 4;
  const half_t* p = base + (long)row * rowStride + kb;
  V16U u;
  u.h[0] = *(const v8h*)(p);
  u.h[1] = *(const v8h*)(p + 8);
  return u.v;
}

// Async copy 16 bytes (8 halves) per lane: global -> LDS, tracked by ASYNCcnt.
// LDS address = low 32 bits of the generic pointer (generic->LDS keeps addr[31:0]).
static __device__ __forceinline__ void async_g2l_b128(half_t* lds, const half_t* g) {
  unsigned loff = (unsigned)(unsigned long long)lds;
  asm volatile("global_load_async_to_lds_b128 %0, %1, off"
               :: "v"(loff), "v"(g) : "memory");
}
static __device__ __forceinline__ void wait_async0() {
  asm volatile("s_wait_asynccnt 0x0" ::: "memory");
}

static __device__ __forceinline__ float sigm(float x) {
  x = fminf(fmaxf(x, -30.f), 30.f);
  return 1.0f / (1.0f + __expf(-x));
}
static __device__ __forceinline__ float tanh_f(float x) {
  x = fminf(fmaxf(x, -15.f), 15.f);
  float e = __expf(2.0f * x);
  return (e - 1.0f) / (e + 1.0f);
}

// One LSTM layer, one timestep.
// gates[64,2048] = A0[64,KT0*32] @ Wx + hprev[64,512] @ Wh + bsum, then cell update.
// Grid: 32 blocks (one 16-wide hidden tile each) x 128 threads (4 waves, one M-tile each).
// Each wave accumulates the 16x16 tiles of ALL FOUR gates so the cell update is in-register.
// B fragments are staged in LDS with async global->LDS copies, double-buffered across kt;
// the ping-pong buffer index carries across the x-phase -> h-phase boundary so the
// prefetch pipeline never drains.
template<int KT0>
__global__ __launch_bounds__(128, 4)
void lstm_step_kernel(const half_t* __restrict__ A0, long a0RowStride, long a0Offset,
                      const half_t* __restrict__ Wx,     // packed [ (g*32+hn)*KT0 + kt ] tiles
                      const half_t* __restrict__ hprev,  // [64,512] f16
                      const half_t* __restrict__ Wh,     // packed, KT=16
                      const float*  __restrict__ bsum,   // [2048] = bih+bhh
                      float*  __restrict__ cstate,       // [64,512] in-place
                      half_t* __restrict__ hnext,        // [64,512] f16
                      float*  __restrict__ outH,         // null or [B,S,H]
                      float*  __restrict__ outC,
                      float*  __restrict__ finH,         // null or [B,H] (written at t==S-1)
                      float*  __restrict__ finC,
                      int t)
{
  __shared__ __align__(128) half_t smem[2][4][512];   // [buf][gate][frag 1KB]

  const int lane = threadIdx.x & 31;
  const int wave = threadIdx.x >> 5;   // M-tile index AND producer gate index
  const int hn   = blockIdx.x;         // hidden tile 0..31
  const unsigned laneB = (unsigned)(lane << 4);

  // producer weight-fragment streams (contiguous in kt: stride 512 halves)
  const half_t* WxP = Wx + (size_t)(wave * 32 + hn) * KT0 * 512;
  const half_t* WhP = Wh + (size_t)(wave * 32 + hn) * 16  * 512;

  v8f acc0 = 0.0f, acc1 = 0.0f, acc2 = 0.0f, acc3 = 0.0f;

  // 4 WMMAs from the LDS-staged B fragments of buffer `bufSel`
  auto mma4 = [&](v16h a, int bufSel) {
    const half_t* bb = &smem[bufSel][0][laneB];
    v16h b0 = *(const v16h*)(bb);
    acc0 = __builtin_amdgcn_wmma_f32_16x16x32_f16(false, a, false, b0, (short)0, acc0, false, false);
    v16h b1 = *(const v16h*)(bb + 512);
    acc1 = __builtin_amdgcn_wmma_f32_16x16x32_f16(false, a, false, b1, (short)0, acc1, false, false);
    v16h b2 = *(const v16h*)(bb + 1024);
    acc2 = __builtin_amdgcn_wmma_f32_16x16x32_f16(false, a, false, b2, (short)0, acc2, false, false);
    v16h b3 = *(const v16h*)(bb + 1536);
    acc3 = __builtin_amdgcn_wmma_f32_16x16x32_f16(false, a, false, b3, (short)0, acc3, false, false);
  };
  auto stage = [&](const half_t* src, int bufSel) {
    half_t* d = &smem[bufSel][wave][laneB];
    const half_t* s = src + laneB;
    async_g2l_b128(d, s);
    async_g2l_b128(d + 8, s + 8);
  };

  // ---- prologue: stage kt=0 into buffer 0 ----
  stage(WxP, 0);
  wait_async0();
  __syncthreads();

  const half_t* aBaseX = A0 + (long)wave * 16 * a0RowStride + a0Offset;
  const half_t* aBaseH = hprev + (long)wave * 16 * H_;

  int buf = 0;
  // ---- phase 1: input GEMM, K = KT0*32 ----
  #pragma unroll 1
  for (int kt = 0; kt < KT0; ++kt) {
    const int j = kt + 1;                            // next k-tile (x phase or first h tile)
    stage((j < KT0) ? (WxP + (size_t)j * 512) : WhP, buf ^ 1);
    v16h a = load_a_frag(aBaseX + kt * 32, a0RowStride, lane);
    mma4(a, buf);
    wait_async0();      // our prefetch landed in LDS
    __syncthreads();    // everyone done reading buf, everyone's prefetch visible
    buf ^= 1;
  }
  // ---- phase 2: recurrent GEMM, K = 512 ----
  #pragma unroll 1
  for (int kt = 0; kt < 16; ++kt) {
    const int j = kt + 1;
    if (j < 16) stage(WhP + (size_t)j * 512, buf ^ 1);
    v16h a = load_a_frag(aBaseH + kt * 32, H_, lane);
    mma4(a, buf);
    wait_async0();
    __syncthreads();
    buf ^= 1;
  }

  // ---- cell update, in-register ----
  // C/D layout: lanes 0-15 N=lane M=r ; lanes 16-31 N=lane-16 M=8+r
  const int col   = lane & 15;
  const int hidx  = hn * 16 + col;
  const int rbase = (lane >> 4) << 3;
  const float bi = bsum[0 * H_ + hidx];
  const float bf = bsum[1 * H_ + hidx];
  const float bg = bsum[2 * H_ + hidx];
  const float bo = bsum[3 * H_ + hidx];

  #pragma unroll
  for (int r = 0; r < 8; ++r) {
    const int m   = wave * 16 + rbase + r;
    const int idx = m * H_ + hidx;
    const float iv = sigm(acc0[r] + bi);
    const float fv = sigm(acc1[r] + bf);
    const float gv = tanh_f(acc2[r] + bg);
    const float ov = sigm(acc3[r] + bo);
    const float cn = fv * cstate[idx] + iv * gv;
    const float hv = ov * tanh_f(cn);
    cstate[idx] = cn;
    hnext[idx]  = (half_t)hv;
    if (outH) {
      const size_t o = ((size_t)m * S_ + (size_t)t) * H_ + hidx;
      outH[o] = hv;
      outC[o] = cn;
    }
    if (finH && t == S_ - 1) { finH[idx] = hv; finC[idx] = cn; }
  }
}

// ---- packing kernels (run once per call; negligible vs 2048 step kernels) ----

// input f32 [B,S,300] -> f16 [B,S,320] zero-padded
__global__ void pack_x_f16(const float* __restrict__ x, half_t* __restrict__ xf) {
  int i = blockIdx.x * 256 + threadIdx.x;           // i < B*S*IP_
  int row = i / IP_;
  int k   = i - row * IP_;
  xf[i] = (k < I_) ? (half_t)x[(size_t)row * I_ + k] : (half_t)0.f;
}

// weight f32 [2048, K] -> fragment-major f16, tile = ntile*KT + kt (512 halves/tile)
// element: n = ntile*16 + (lane&15), k = kt*32 + (lane>>4)*16 + h
__global__ void pack_w_frag(const float* __restrict__ W, half_t* __restrict__ Wp,
                            int K, int KT) {
  int idx = blockIdx.x * 256 + threadIdx.x;          // idx < 128*KT*512
  int tile   = idx >> 9;
  int within = idx & 511;
  int lane = within >> 4;
  int h    = within & 15;
  int ntile = tile / KT;
  int kt    = tile - ntile * KT;
  int n = ntile * 16 + (lane & 15);
  int k = kt * 32 + ((lane >> 4) << 4) + h;
  Wp[idx] = (k < K) ? (half_t)W[(size_t)n * K + k] : (half_t)0.f;
}

__global__ void bias_sum_kernel(const float* __restrict__ a, const float* __restrict__ b,
                                float* __restrict__ o, int n) {
  int i = blockIdx.x * 256 + threadIdx.x;
  if (i < n) o[i] = a[i] + b[i];
}

__global__ void zero_f32_kernel(float* p, int n) {
  int i = blockIdx.x * 256 + threadIdx.x;
  if (i < n) p[i] = 0.f;
}
__global__ void zero_f16_kernel(half_t* p, int n) {
  int i = blockIdx.x * 256 + threadIdx.x;
  if (i < n) p[i] = (half_t)0.f;
}

static inline size_t alignUp(size_t v, size_t a) { return (v + a - 1) & ~(a - 1); }

extern "C" void kernel_launch(void* const* d_in, const int* in_sizes, int n_in,
                              void* d_out, int out_size, void* d_ws, size_t ws_size,
                              hipStream_t stream) {
  const float* input = (const float*)d_in[0];
  const float* Wih0  = (const float*)d_in[1];
  const float* Whh0  = (const float*)d_in[2];
  const float* bih0  = (const float*)d_in[3];
  const float* bhh0  = (const float*)d_in[4];
  const float* Wih1  = (const float*)d_in[5];
  const float* Whh1  = (const float*)d_in[6];
  const float* bih1  = (const float*)d_in[7];
  const float* bhh1  = (const float*)d_in[8];

  // ---- workspace layout (bump allocator, 256B aligned) ----
  char* ws = (char*)d_ws;
  size_t off = 0;
  auto take = [&](size_t bytes) { char* p = ws + off; off = alignUp(off + bytes, 256); return p; };

  const int KT_X = IP_ / 32;  // 10
  const int KT_H = H_ / 32;   // 16

  half_t* xf16  = (half_t*)take((size_t)B_ * S_ * IP_ * 2);
  half_t* Wih0p = (half_t*)take((size_t)128 * KT_X * 512 * 2);
  half_t* Whh0p = (half_t*)take((size_t)128 * KT_H * 512 * 2);
  half_t* Wih1p = (half_t*)take((size_t)128 * KT_H * 512 * 2);
  half_t* Whh1p = (half_t*)take((size_t)128 * KT_H * 512 * 2);
  float*  bsum0 = (float*)take((size_t)G_ * 4);
  float*  bsum1 = (float*)take((size_t)G_ * 4);
  half_t* h0buf = (half_t*)take((size_t)2 * B_ * H_ * 2);   // ping-pong
  half_t* h1buf = (half_t*)take((size_t)2 * B_ * H_ * 2);
  float*  c0    = (float*)take((size_t)B_ * H_ * 4);
  float*  c1    = (float*)take((size_t)B_ * H_ * 4);
  if (off > ws_size) return;  // workspace too small: do nothing (deterministic)

  // ---- output layout: concat(out[B,S,H], c_out[B,S,H], h1[B,H], c1[B,H]) ----
  float* outH = (float*)d_out;
  float* outC = outH + (size_t)B_ * S_ * H_;
  float* finH = outC + (size_t)B_ * S_ * H_;
  float* finC = finH + (size_t)B_ * H_;

  // ---- pack & init ----
  pack_x_f16<<<(B_ * S_ * IP_) / 256, 256, 0, stream>>>(input, xf16);
  pack_w_frag<<<(128 * KT_X * 512) / 256, 256, 0, stream>>>(Wih0, Wih0p, I_, KT_X);
  pack_w_frag<<<(128 * KT_H * 512) / 256, 256, 0, stream>>>(Whh0, Whh0p, H_, KT_H);
  pack_w_frag<<<(128 * KT_H * 512) / 256, 256, 0, stream>>>(Wih1, Wih1p, H_, KT_H);
  pack_w_frag<<<(128 * KT_H * 512) / 256, 256, 0, stream>>>(Whh1, Whh1p, H_, KT_H);
  bias_sum_kernel<<<(G_ + 255) / 256, 256, 0, stream>>>(bih0, bhh0, bsum0, G_);
  bias_sum_kernel<<<(G_ + 255) / 256, 256, 0, stream>>>(bih1, bhh1, bsum1, G_);
  zero_f16_kernel<<<(2 * B_ * H_ + 255) / 256, 256, 0, stream>>>(h0buf, 2 * B_ * H_);
  zero_f16_kernel<<<(2 * B_ * H_ + 255) / 256, 256, 0, stream>>>(h1buf, 2 * B_ * H_);
  zero_f32_kernel<<<(B_ * H_ + 255) / 256, 256, 0, stream>>>(c0, B_ * H_);
  zero_f32_kernel<<<(B_ * H_ + 255) / 256, 256, 0, stream>>>(c1, B_ * H_);

  // ---- recurrent chain: 2 kernels per step, sync via launch boundaries ----
  for (int t = 0; t < S_; ++t) {
    const half_t* h0r = h0buf + (size_t)(t & 1) * B_ * H_;
    half_t*       h0w = h0buf + (size_t)((t + 1) & 1) * B_ * H_;
    const half_t* h1r = h1buf + (size_t)(t & 1) * B_ * H_;
    half_t*       h1w = h1buf + (size_t)((t + 1) & 1) * B_ * H_;

    // layer 0: A = x_t (rows stride S*IP, offset t*IP), K = 320 + 512
    lstm_step_kernel<KT_X><<<32, 128, 0, stream>>>(
        xf16, (long)S_ * IP_, (long)t * IP_,
        Wih0p, h0r, Whh0p, bsum0, c0, h0w,
        nullptr, nullptr, nullptr, nullptr, t);

    // layer 1: A = h0_new (rows stride 512), K = 512 + 512
    lstm_step_kernel<KT_H><<<32, 128, 0, stream>>>(
        h0w, (long)H_, 0L,
        Wih1p, h1r, Whh1p, bsum1, c1, h1w,
        outH, outC, finH, finC, t);
  }
}
